// FlaxAttentionBlock_85495618994429
// MI455X (gfx1250) — matrix operand order repaired and checked by reading, not compiled, not run on hardware
//
#include <hip/hip_runtime.h>

typedef __bf16 bf16;
typedef __attribute__((ext_vector_type(16))) __bf16 v16bf;
typedef __attribute__((ext_vector_type(8)))  float  v8f;
typedef __attribute__((ext_vector_type(4)))  unsigned u32x4;
typedef __attribute__((ext_vector_type(8)))  int      i32x8;
typedef __attribute__((ext_vector_type(4)))  int      i32x4;

#define HEADS    8
#define DIM_HEAD 64
#define INNER    512   // HEADS*DIM_HEAD
#define SEQ      2048
#define BATCH    4
#define DMODEL   512

#if __has_builtin(__builtin_amdgcn_tensor_load_to_lds)
#define HAVE_TDM 1
#else
#define HAVE_TDM 0
#endif

// ---------------------------------------------------------------------------
// CDNA5 async global->LDS copy (16B per lane), tracked by ASYNCcnt.
// Syntax modeled on compiler output: global_load_b128 vdst, v[a:a+1], off
// Here VDST carries the LDS byte address (ISA 15.18.3 op 98).
// ---------------------------------------------------------------------------
__device__ inline void async_ld16(void* lptr, const bf16* gptr) {
  unsigned l = (unsigned)(uintptr_t)lptr;          // low 32 bits = LDS offset
  unsigned long long g = (unsigned long long)(uintptr_t)gptr;
  asm volatile("global_load_async_to_lds_b128 %0, %1, off"
               :: "v"(l), "v"(g) : "memory");
}
__device__ inline void async_wait0() {
  asm volatile("s_wait_asynccnt 0x0" ::: "memory");
}

// ---------------------------------------------------------------------------
// Tensor Data Mover: 2D tile (rows x cols bf16 elems) from global to LDS.
// D# group0: count=1, lds_addr, global_addr[56:0], type=2.
// D# group1: data_size=2B, tensor_dim0=cols, tensor_dim1=rows,
//            tile_dim0=cols, tile_dim1=rows, dim0_stride=row_stride.
// This toolchain exposes the 6-arg builtin:
//   (u32x4 g0, i32x8 g1, i32x4 g2, i32x4 g3, i32x8 extra, i32 cpol)
// ---------------------------------------------------------------------------
#if HAVE_TDM
__device__ inline void tdm_load_2d(unsigned lds_addr, const bf16* gptr,
                                   int rows, int cols, int row_stride_elems) {
  unsigned long long ga = (unsigned long long)(uintptr_t)gptr;
  u32x4 g0;
  g0.x = 1u;                                                // count=1
  g0.y = lds_addr;                                          // lds_addr
  g0.z = (unsigned)ga;                                      // gaddr[31:0]
  g0.w = (unsigned)((ga >> 32) & 0x01ffffffu) | (2u << 30); // gaddr hi | type=2
  i32x8 g1;
  g1[0] = (int)(1u << 16);                                  // data_size=2B
  g1[1] = (int)(((unsigned)cols & 0xffffu) << 16);          // tdim0 lo16
  g1[2] = (int)((((unsigned)cols) >> 16) |
                (((unsigned)rows & 0xffffu) << 16));        // tdim0 hi | tdim1 lo
  g1[3] = (int)((((unsigned)rows) >> 16) |
                (((unsigned)cols) << 16));                  // tdim1 hi | tile_dim0
  g1[4] = rows;                                             // tile_dim1 (dim2=0)
  g1[5] = row_stride_elems;                                 // dim0_stride lo32
  g1[6] = 0;
  g1[7] = 0;
  i32x4 z4 = {0, 0, 0, 0};
  i32x8 z8 = {0, 0, 0, 0, 0, 0, 0, 0};
  __builtin_amdgcn_tensor_load_to_lds(g0, g1, z4, z4, z8, 0);
}
__device__ inline void tdm_wait0() {
#if __has_builtin(__builtin_amdgcn_s_wait_tensorcnt)
  __builtin_amdgcn_s_wait_tensorcnt(0);
#else
  asm volatile("s_wait_tensorcnt 0x0" ::: "memory");
#endif
}
#endif

// ---------------------------------------------------------------------------
// fp32 -> bf16 conversion (grid-stride)
// ---------------------------------------------------------------------------
__global__ void cvt_f32_to_bf16(const float* __restrict__ src,
                                bf16* __restrict__ dst, int n) {
  int i = blockIdx.x * blockDim.x + threadIdx.x;
  int stride = gridDim.x * blockDim.x;
  for (; i < n; i += stride) dst[i] = (bf16)src[i];
}

// ---------------------------------------------------------------------------
// WMMA fragment loaders (wave32, 16x16x32 bf16; contiguous k = 16*half + e in both A and B)
// ---------------------------------------------------------------------------
__device__ inline v16bf lds_load_a_frag(const bf16* tile, int ld) {
  const int lane = threadIdx.x & 31;
  const int m    = lane & 15;
  const int half = lane >> 4;
  v16bf a;
#pragma unroll
  for (int e = 0; e < 8; ++e) a[e]     = tile[m * ld + half * 16 + e];
#pragma unroll
  for (int e = 0; e < 8; ++e) a[8 + e] = tile[m * ld + half * 16 + 8 + e];
  return a;
}

__device__ inline v16bf lds_load_b_frag(const bf16* tile, int ld, int n0) {
  const int lane = threadIdx.x & 31;
  const int n    = (lane & 15) + n0;
  const int kb   = (lane >> 4) * 16;
  v16bf b;
#pragma unroll
  for (int e = 0; e < 16; ++e) b[e] = tile[(kb + e) * ld + n];
  return b;
}

// ---------------------------------------------------------------------------
// bf16 WMMA GEMM: C[MxN] = A[MxK]*B[KxN] (+bias). Block tile 256x64, BK=32,
// 256 threads = 8 waves; wave owns 32 rows x 64 cols = 8 WMMA tiles, so each
// staged B fragment feeds two WMMAs. A/B staged via async global->LDS.
// ---------------------------------------------------------------------------
template <bool OUT_BF16, bool HAS_BIAS>
__global__ __launch_bounds__(256) void gemm_bf16_wmma(
    const bf16* __restrict__ A, const bf16* __restrict__ B,
    void* __restrict__ C, const float* __restrict__ bias,
    int M, int N, int K) {
  __shared__ __align__(16) bf16 As[256 * 32];  // 16 KB
  __shared__ __align__(16) bf16 Bs[32 * 64];   // 4 KB

  const int bm   = blockIdx.x * 256;
  const int bn   = blockIdx.y * 64;
  const int tid  = threadIdx.x;
  const int wave = tid >> 5;
  const int lane = tid & 31;

  v8f acc[2][4] = {};

  for (int k0 = 0; k0 < K; k0 += 32) {
    // Stage A tile (256x32 bf16 = 16KB): 1024 16B-chunks, 4 per thread.
#pragma unroll
    for (int i = 0; i < 4; ++i) {
      int idx = tid + i * 256;
      int r = idx >> 2;
      int c = (idx & 3) * 8;
      async_ld16(&As[r * 32 + c], &A[(size_t)(bm + r) * K + k0 + c]);
    }
    // Stage B tile (32x64 bf16 = 4KB): 256 16B-chunks, 1 per thread.
    {
      int r = tid >> 3;
      int c = (tid & 7) * 8;
      async_ld16(&Bs[r * 64 + c], &B[(size_t)(k0 + r) * N + bn + c]);
    }
    if (k0 + 32 < K)  // hint next A K-slab toward L2 (global_prefetch_b8)
      __builtin_prefetch(&A[(size_t)(bm + (tid >> 2)) * K + k0 + 32], 0, 1);
    async_wait0();
    __syncthreads();

    v16bf a0 = lds_load_a_frag(&As[(wave * 32 + 0) * 32], 32);
    v16bf a1 = lds_load_a_frag(&As[(wave * 32 + 16) * 32], 32);
#pragma unroll
    for (int nt = 0; nt < 4; ++nt) {
      v16bf b = lds_load_b_frag(Bs, 64, nt * 16);
      acc[0][nt] = __builtin_amdgcn_wmma_f32_16x16x32_bf16(
          false, a0, false, b, (short)0, acc[0][nt], false, false);
      acc[1][nt] = __builtin_amdgcn_wmma_f32_16x16x32_bf16(
          false, a1, false, b, (short)0, acc[1][nt], false, false);
    }
    __syncthreads();
  }

  const int half = lane >> 4;
  const int n_in = lane & 15;
#pragma unroll
  for (int mt = 0; mt < 2; ++mt) {
#pragma unroll
    for (int nt = 0; nt < 4; ++nt) {
#pragma unroll
      for (int r = 0; r < 8; ++r) {
        int m = wave * 32 + mt * 16 + r + 8 * half;
        int n = bn + nt * 16 + n_in;
        float v = acc[mt][nt][r];
        if (HAS_BIAS) v += bias[n];
        if (OUT_BF16)
          ((bf16*)C)[(size_t)(bm + m) * N + n] = (bf16)v;
        else
          ((float*)C)[(size_t)(bm + m) * N + n] = v;
      }
    }
  }
}

// ---------------------------------------------------------------------------
// Flash-style attention. Grid: (SEQ/64, HEADS, BATCH), block = 128 (4 waves).
// Wave owns 16 query rows x dh=64. KV tile (32 keys) staged by the TDM.
// ---------------------------------------------------------------------------
__global__ __launch_bounds__(128) void attn_wmma(
    const bf16* __restrict__ Q, const bf16* __restrict__ KV,
    bf16* __restrict__ O) {
  __shared__ __align__(16) bf16 Ks[32 * 64];
  __shared__ __align__(16) bf16 Vs[32 * 64];
  __shared__ float Ssc[4][16 * 32];
  __shared__ bf16  Pb[4][16 * 32];
  __shared__ float Sal[4][16];

  const int wave = threadIdx.x >> 5;
  const int lane = threadIdx.x & 31;
  const int bidx = blockIdx.z;
  const int h    = blockIdx.y;
  const int q0   = blockIdx.x * 64 + wave * 16;
  const float scale = 0.125f;  // 1/sqrt(64), exact in bf16

  // Q fragments (A layout), two dh-chunks of 32.
  v16bf qa[2];
  {
    const int m    = lane & 15;
    const int half = lane >> 4;
    const bf16* Qb = Q + ((size_t)bidx * SEQ + q0 + m) * INNER + h * DIM_HEAD;
#pragma unroll
    for (int c = 0; c < 2; ++c) {
#pragma unroll
      for (int e = 0; e < 8; ++e) qa[c][e]     = Qb[c * 32 + half * 16 + e];
#pragma unroll
      for (int e = 0; e < 8; ++e) qa[c][8 + e] = Qb[c * 32 + half * 16 + 8 + e];
    }
  }

  float m_i = -3.0e38f, l_i = 0.0f;
  v8f o[4] = {};

  for (int j = 0; j < SEQ; j += 32) {
    // ---- Stage K and V tiles (32 keys x 64 dh bf16 = 4KB each) ----
#if HAVE_TDM
    if (threadIdx.x < 32) {  // wave 0 drives the Tensor Data Mover
      const bf16* kg =
          KV + ((size_t)bidx * SEQ + j) * (size_t)(2 * INNER) + h * DIM_HEAD;
      tdm_load_2d((unsigned)(uintptr_t)Ks, kg,        32, 64, 2 * INNER);
      tdm_load_2d((unsigned)(uintptr_t)Vs, kg + INNER, 32, 64, 2 * INNER);
      tdm_wait0();
    }
    __syncthreads();
#else
#pragma unroll
    for (int i = 0; i < 2; ++i) {
      int idx = (int)threadIdx.x + i * 128;
      int r = idx >> 3;
      int c = (idx & 7) * 8;
      size_t g = ((size_t)bidx * SEQ + j + r) * (size_t)(2 * INNER) +
                 h * DIM_HEAD + c;
      *(uint4*)&Ks[r * 64 + c] = *(const uint4*)&KV[g];
      *(uint4*)&Vs[r * 64 + c] = *(const uint4*)&KV[g + INNER];
    }
    __syncthreads();
#endif

    // ---- Scores S = Q*K^T (16 q-rows x 32 keys) ----
#pragma unroll
    for (int g = 0; g < 2; ++g) {
      v8f s = {};
#pragma unroll
      for (int c = 0; c < 2; ++c) {
        v16bf kb;
        const int n   = lane & 15;
        const int kb0 = (lane >> 4) * 16;
#pragma unroll
        for (int e = 0; e < 16; ++e)
          kb[e] = Ks[(g * 16 + n) * 64 + c * 32 + kb0 + e];
        s = __builtin_amdgcn_wmma_f32_16x16x32_bf16(
            false, qa[c], false, kb, (short)0, s, false, false);
      }
      const int half = lane >> 4;
#pragma unroll
      for (int r = 0; r < 8; ++r)
        Ssc[wave][(r + 8 * half) * 32 + g * 16 + (lane & 15)] = s[r];
    }
    __syncthreads();

    // ---- Online softmax: lanes 0..15 each own one query row ----
    if (lane < 16) {
      float* row = &Ssc[wave][lane * 32];
      float mx = m_i;
#pragma unroll
      for (int i = 0; i < 32; ++i) mx = fmaxf(mx, row[i] * scale);
      float alpha = __expf(m_i - mx);
      float sum = 0.0f;
#pragma unroll
      for (int i = 0; i < 32; ++i) {
        float p = __expf(row[i] * scale - mx);
        sum += p;
        Pb[wave][lane * 32 + i] = (bf16)p;  // bf16 probs, matches reference
      }
      m_i = mx;
      l_i = l_i * alpha + sum;
      Sal[wave][lane] = alpha;
    }
    __syncthreads();

    // ---- Rescale accumulators, then O += P*V (4 dh N-tiles) ----
    {
      const int half = lane >> 4;
#pragma unroll
      for (int r = 0; r < 8; ++r) {
        float al = Sal[wave][r + 8 * half];
#pragma unroll
        for (int nt = 0; nt < 4; ++nt) o[nt][r] *= al;
      }
      v16bf pa = lds_load_a_frag(Pb[wave], 32);
#pragma unroll
      for (int nt = 0; nt < 4; ++nt) {
        v16bf vb;
        const int n  = lane & 15;
        const int k0 = (lane >> 4) * 16;
#pragma unroll
        for (int e = 0; e < 16; ++e)
          vb[e] = Vs[(k0 + e) * 64 + nt * 16 + n];
        o[nt] = __builtin_amdgcn_wmma_f32_16x16x32_bf16(
            false, pa, false, vb, (short)0, o[nt], false, false);
      }
    }
    __syncthreads();
  }

  // ---- Normalize and store bf16 attention output [B,S,H,dh] ----
  if (lane < 16) Sal[wave][lane] = 1.0f / l_i;
  __syncthreads();
  {
    const int half = lane >> 4;
    const int n_in = lane & 15;
#pragma unroll
    for (int nt = 0; nt < 4; ++nt) {
#pragma unroll
      for (int r = 0; r < 8; ++r) {
        int m = r + 8 * half;
        float v = o[nt][r] * Sal[wave][m];
        O[((size_t)bidx * SEQ + q0 + m) * INNER + h * DIM_HEAD + nt * 16 +
          n_in] = (bf16)v;
      }
    }
  }
}

// ---------------------------------------------------------------------------
// Host launcher
// ---------------------------------------------------------------------------
extern "C" void kernel_launch(void* const* d_in, const int* in_sizes, int n_in,
                              void* d_out, int out_size, void* d_ws,
                              size_t ws_size, hipStream_t stream) {
  (void)in_sizes; (void)n_in; (void)out_size; (void)ws_size;
  const float* X   = (const float*)d_in[0];
  const float* Wq  = (const float*)d_in[1];
  const float* Wkv = (const float*)d_in[2];
  const float* Wo  = (const float*)d_in[3];
  const float* bo  = (const float*)d_in[4];

  const size_t nX = (size_t)BATCH * SEQ * DMODEL;  // 4,194,304

  char* ws = (char*)d_ws;
  bf16* Xbf   = (bf16*)ws;  ws += nX * 2;
  bf16* Wqbf  = (bf16*)ws;  ws += (size_t)DMODEL * INNER * 2;
  bf16* Wkvbf = (bf16*)ws;  ws += (size_t)DMODEL * 2 * INNER * 2;
  bf16* Wobf  = (bf16*)ws;  ws += (size_t)INNER * DMODEL * 2;
  bf16* Qbf   = (bf16*)ws;  ws += nX * 2;
  bf16* KVbf  = (bf16*)ws;  ws += nX * 2 * 2;
  bf16* Abf   = (bf16*)ws;  ws += nX * 2;

  cvt_f32_to_bf16<<<1024, 256, 0, stream>>>(X, Xbf, (int)nX);
  cvt_f32_to_bf16<<<256, 256, 0, stream>>>(Wq, Wqbf, DMODEL * INNER);
  cvt_f32_to_bf16<<<256, 256, 0, stream>>>(Wkv, Wkvbf, DMODEL * 2 * INNER);
  cvt_f32_to_bf16<<<256, 256, 0, stream>>>(Wo, Wobf, INNER * DMODEL);

  const int M = BATCH * SEQ;  // 8192

  gemm_bf16_wmma<true, false><<<dim3(M / 256, INNER / 64), 256, 0, stream>>>(
      Xbf, Wqbf, Qbf, nullptr, M, INNER, DMODEL);
  gemm_bf16_wmma<true, false>
      <<<dim3(M / 256, (2 * INNER) / 64), 256, 0, stream>>>(
          Xbf, Wkvbf, KVbf, nullptr, M, 2 * INNER, DMODEL);
  attn_wmma<<<dim3(SEQ / 64, HEADS, BATCH), 128, 0, stream>>>(Qbf, KVbf, Abf);
  gemm_bf16_wmma<false, true><<<dim3(M / 256, DMODEL / 64), 256, 0, stream>>>(
      Abf, Wobf, d_out, bo, M, DMODEL, INNER);
}
